// GNN_9371618639981
// MI455X (gfx1250) — compile-verified, matrix operand order checked by reading, not run on hardware
//
#include <hip/hip_runtime.h>
#include <hip/hip_bf16.h>

#define N_NODES 50000
#define N_EDGES 400000
#define IN_DIM  16
#define HID     64
#define EDGE_IN 4
#define N_LAYERS 4

#define K1_PAD 160   // 2*HID + EDGE_IN = 132 -> padded to multiple of 32
#define K1_REAL 132

typedef __attribute__((ext_vector_type(16))) _Float16 v16h;
typedef __attribute__((ext_vector_type(8)))  _Float16 v8h;
typedef __attribute__((ext_vector_type(8)))  float    v8f;

__device__ __forceinline__ v16h lds_frag16(const _Float16* p) {
    // WMMA 16-bit A/B fragment: this lane's 16 halves are K {kb..kb+7} and
    // {kb+16..kb+23} -> two contiguous 16B LDS loads.
    v8h lo = *(const v8h*)p;
    v8h hb = *(const v8h*)(p + 16);
    v16h r;
#pragma unroll
    for (int i = 0; i < 8; ++i) { r[i] = lo[i]; r[8 + i] = hb[i]; }
    return r;
}

__device__ __forceinline__ float silu_f(float x) {
    return x / (1.0f + __expf(-x));
}

// ---------------- weight prep: f32 -> f16, transposed [n][k], K padded ------

__global__ void prep_edge_w1(const float* __restrict__ w, _Float16* __restrict__ wt) {
    int i = blockIdx.x * blockDim.x + threadIdx.x;       // [L][64][K1_PAD]
    if (i >= N_LAYERS * HID * K1_PAD) return;
    int k = i % K1_PAD;
    int n = (i / K1_PAD) % HID;
    int l = i / (K1_PAD * HID);
    float v = (k < K1_REAL) ? w[((size_t)l * K1_REAL + k) * HID + n] : 0.0f;
    wt[i] = (_Float16)v;
}

__global__ void prep_node_w1(const float* __restrict__ w, _Float16* __restrict__ wt) {
    int i = blockIdx.x * blockDim.x + threadIdx.x;       // [L][64][128]
    if (i >= N_LAYERS * HID * 2 * HID) return;
    int k = i % (2 * HID);
    int n = (i / (2 * HID)) % HID;
    int l = i / (2 * HID * HID);
    wt[i] = (_Float16)w[((size_t)l * 2 * HID + k) * HID + n];
}

__global__ void prep_sq(const float* __restrict__ w, _Float16* __restrict__ wt) {
    int i = blockIdx.x * blockDim.x + threadIdx.x;       // [L][64][64] transpose
    if (i >= N_LAYERS * HID * HID) return;
    int k = i % HID;
    int n = (i / HID) % HID;
    int l = i / (HID * HID);
    wt[i] = (_Float16)w[((size_t)l * HID + k) * HID + n];
}

// ---------------- embedding: h = nodes @ emb_w + emb_b ----------------------

__global__ void embed_kernel(const float* __restrict__ nodes,
                             const float* __restrict__ emb_w,
                             const float* __restrict__ emb_b,
                             float* __restrict__ h) {
    int gid = blockIdx.x * blockDim.x + threadIdx.x;
    if (gid >= N_NODES * HID) return;
    int node = gid >> 6, j = gid & 63;
    const float* nr = nodes + (size_t)node * IN_DIM;
    float acc = emb_b[j];
#pragma unroll
    for (int k = 0; k < IN_DIM; ++k) acc += nr[k] * emb_w[k * HID + j];
    h[gid] = acc;
}

// ---------------- edge model: gather -> MLP(WMMA) -> mask -> scatter-add ----

__global__ __launch_bounds__(128) void edge_kernel(
        const float* __restrict__ hin, const int* __restrict__ eidx,
        const float* __restrict__ emask, const float* __restrict__ eattr,
        const _Float16* __restrict__ w1t, const float* __restrict__ b1,
        const _Float16* __restrict__ w2t, const float* __restrict__ b2,
        float* __restrict__ agg) {
    __shared__ __align__(16) _Float16 sA[64][K1_PAD];     // e_in tile (f16)
    __shared__ __align__(16) _Float16 sW1[HID][K1_PAD];   // w1 transposed [n][k]
    __shared__ __align__(16) _Float16 sE[64][HID];        // hidden tile
    __shared__ __align__(16) _Float16 sW2[HID][HID];      // w2 transposed
    __shared__ float sMask[64];
    __shared__ int   sRow[64];

    const int tid = threadIdx.x;
    const int eb  = blockIdx.x * 64;

    // --- stage e_in = [h[row] | h[col] | edge_attr | 0-pad] into LDS (f16) ---
    {
        int el = tid >> 1, half = tid & 1;
        int e = eb + el;
        if (half == 0) {
            int r = eidx[e];
            sRow[el] = r;
            const float4* src = (const float4*)(hin + (size_t)r * HID);
#pragma unroll
            for (int j = 0; j < 16; ++j) {
                float4 v = src[j];
                _Float16* d = &sA[el][j * 4];
                d[0] = (_Float16)v.x; d[1] = (_Float16)v.y;
                d[2] = (_Float16)v.z; d[3] = (_Float16)v.w;
            }
        } else {
            int c = eidx[N_EDGES + e];
            sMask[el] = emask[e];
            const float4* src = (const float4*)(hin + (size_t)c * HID);
#pragma unroll
            for (int j = 0; j < 16; ++j) {
                float4 v = src[j];
                _Float16* d = &sA[el][HID + j * 4];
                d[0] = (_Float16)v.x; d[1] = (_Float16)v.y;
                d[2] = (_Float16)v.z; d[3] = (_Float16)v.w;
            }
            float4 a = *(const float4*)(eattr + (size_t)e * EDGE_IN);
            _Float16* d = &sA[el][2 * HID];
            d[0] = (_Float16)a.x; d[1] = (_Float16)a.y;
            d[2] = (_Float16)a.z; d[3] = (_Float16)a.w;
#pragma unroll
            for (int k = K1_REAL; k < K1_PAD; ++k) sA[el][k] = (_Float16)0.0f;
        }
    }
    // --- stage weights ---
    {
        const uint4* s1 = (const uint4*)w1t;
        uint4* d1 = (uint4*)&sW1[0][0];
        for (int i = tid; i < HID * K1_PAD / 8; i += 128) d1[i] = s1[i];
        const uint4* s2 = (const uint4*)w2t;
        uint4* d2 = (uint4*)&sW2[0][0];
        for (int i = tid; i < HID * HID / 8; i += 128) d2[i] = s2[i];
    }
    __syncthreads();

    const int wave = tid >> 5, lane = tid & 31;
    const int ln = lane & 15, hb = (lane >> 4) * 8;   // K sub-block offset
    const int m = wave * 16 + ln;                     // A-fragment row

    // --- GEMM1: e = silu(e_in @ w1 + b1) ---
#pragma unroll
    for (int nt = 0; nt < 4; ++nt) {
        float bias = b1[nt * 16 + ln];
        v8f c;
#pragma unroll
        for (int r = 0; r < 8; ++r) c[r] = bias;
#pragma unroll
        for (int kt = 0; kt < K1_PAD / 32; ++kt) {
            v16h a = lds_frag16(&sA[m][kt * 32 + hb]);
            v16h b = lds_frag16(&sW1[nt * 16 + ln][kt * 32 + hb]);
            c = __builtin_amdgcn_wmma_f32_16x16x32_f16(false, a, false, b,
                                                       (short)0, c, false, false);
        }
#pragma unroll
        for (int r = 0; r < 8; ++r)
            sE[wave * 16 + r + hb][nt * 16 + ln] = (_Float16)silu_f(c[r]);
    }
    __syncthreads();

    // --- GEMM2: e = silu(e @ w2 + b2) * mask ; scatter-add into agg ---
#pragma unroll
    for (int nt = 0; nt < 4; ++nt) {
        float bias = b2[nt * 16 + ln];
        v8f c;
#pragma unroll
        for (int r = 0; r < 8; ++r) c[r] = bias;
#pragma unroll
        for (int kt = 0; kt < HID / 32; ++kt) {
            v16h a = lds_frag16(&sE[m][kt * 32 + hb]);
            v16h b = lds_frag16(&sW2[nt * 16 + ln][kt * 32 + hb]);
            c = __builtin_amdgcn_wmma_f32_16x16x32_f16(false, a, false, b,
                                                       (short)0, c, false, false);
        }
#pragma unroll
        for (int r = 0; r < 8; ++r) {
            int erow = wave * 16 + r + hb;                 // local edge (D row)
            float v = silu_f(c[r]) * sMask[erow];
            int dst = sRow[erow];
            unsafeAtomicAdd(&agg[(size_t)dst * HID + nt * 16 + ln], v);
        }
    }
}

// ---------------- node model: concat(h,agg) -> 2-layer MLP ------------------

__global__ __launch_bounds__(128) void node_kernel(
        const float* __restrict__ hin, const float* __restrict__ agg,
        const _Float16* __restrict__ w1t, const float* __restrict__ b1,
        const _Float16* __restrict__ w2t, const float* __restrict__ b2,
        float* __restrict__ hout) {
    __shared__ __align__(16) _Float16 sA[64][2 * HID];
    __shared__ __align__(16) _Float16 sW1[HID][2 * HID];
    __shared__ __align__(16) _Float16 sE[64][HID];
    __shared__ __align__(16) _Float16 sW2[HID][HID];

    const int tid = threadIdx.x;
    const int nb  = blockIdx.x * 64;

    {
        int el = tid >> 1, half = tid & 1;
        int node = nb + el;
        bool valid = node < N_NODES;
        const float* src_f = half ? agg : hin;
        _Float16* d0 = &sA[el][half * HID];
        if (valid) {
            const float4* src = (const float4*)(src_f + (size_t)node * HID);
#pragma unroll
            for (int j = 0; j < 16; ++j) {
                float4 v = src[j];
                _Float16* d = d0 + j * 4;
                d[0] = (_Float16)v.x; d[1] = (_Float16)v.y;
                d[2] = (_Float16)v.z; d[3] = (_Float16)v.w;
            }
        } else {
#pragma unroll
            for (int k = 0; k < HID; ++k) d0[k] = (_Float16)0.0f;
        }
    }
    {
        const uint4* s1 = (const uint4*)w1t;
        uint4* d1 = (uint4*)&sW1[0][0];
        for (int i = tid; i < HID * 2 * HID / 8; i += 128) d1[i] = s1[i];
        const uint4* s2 = (const uint4*)w2t;
        uint4* d2 = (uint4*)&sW2[0][0];
        for (int i = tid; i < HID * HID / 8; i += 128) d2[i] = s2[i];
    }
    __syncthreads();

    const int wave = tid >> 5, lane = tid & 31;
    const int ln = lane & 15, hb = (lane >> 4) * 8;
    const int m = wave * 16 + ln;

#pragma unroll
    for (int nt = 0; nt < 4; ++nt) {
        float bias = b1[nt * 16 + ln];
        v8f c;
#pragma unroll
        for (int r = 0; r < 8; ++r) c[r] = bias;
#pragma unroll
        for (int kt = 0; kt < 2 * HID / 32; ++kt) {
            v16h a = lds_frag16(&sA[m][kt * 32 + hb]);
            v16h b = lds_frag16(&sW1[nt * 16 + ln][kt * 32 + hb]);
            c = __builtin_amdgcn_wmma_f32_16x16x32_f16(false, a, false, b,
                                                       (short)0, c, false, false);
        }
#pragma unroll
        for (int r = 0; r < 8; ++r)
            sE[wave * 16 + r + hb][nt * 16 + ln] = (_Float16)silu_f(c[r]);
    }
    __syncthreads();

#pragma unroll
    for (int nt = 0; nt < 4; ++nt) {
        float bias = b2[nt * 16 + ln];
        v8f c;
#pragma unroll
        for (int r = 0; r < 8; ++r) c[r] = bias;
#pragma unroll
        for (int kt = 0; kt < HID / 32; ++kt) {
            v16h a = lds_frag16(&sE[m][kt * 32 + hb]);
            v16h b = lds_frag16(&sW2[nt * 16 + ln][kt * 32 + hb]);
            c = __builtin_amdgcn_wmma_f32_16x16x32_f16(false, a, false, b,
                                                       (short)0, c, false, false);
        }
#pragma unroll
        for (int r = 0; r < 8; ++r) {
            int node = nb + wave * 16 + r + hb;
            if (node < N_NODES)
                hout[(size_t)node * HID + nt * 16 + ln] = c[r];
        }
    }
}

// ---------------------------- host driver -----------------------------------

extern "C" void kernel_launch(void* const* d_in, const int* in_sizes, int n_in,
                              void* d_out, int out_size, void* d_ws, size_t ws_size,
                              hipStream_t stream) {
    const float* nodes = (const float*)d_in[0];
    const int*   eidx  = (const int*)d_in[1];
    const float* emask = (const float*)d_in[2];
    const float* eattr = (const float*)d_in[3];
    const float* emb_w = (const float*)d_in[4];
    const float* emb_b = (const float*)d_in[5];
    const float* ew1   = (const float*)d_in[6];
    const float* eb1   = (const float*)d_in[7];
    const float* ew2   = (const float*)d_in[8];
    const float* eb2   = (const float*)d_in[9];
    const float* nw1   = (const float*)d_in[10];
    const float* nb1   = (const float*)d_in[11];
    const float* nw2   = (const float*)d_in[12];
    const float* nb2   = (const float*)d_in[13];

    const size_t NH = (size_t)N_NODES * HID;           // floats
    char* ws = (char*)d_ws;
    float* hA  = (float*)(ws);                          // 12.8 MB
    float* hB  = (float*)(ws + NH * 4);                 // 12.8 MB
    float* agg = (float*)(ws + 2 * NH * 4);             // 12.8 MB
    size_t woff = 3 * NH * 4;
    _Float16* we1t = (_Float16*)(ws + woff);  woff += (size_t)N_LAYERS * HID * K1_PAD * 2;
    _Float16* we2t = (_Float16*)(ws + woff);  woff += (size_t)N_LAYERS * HID * HID * 2;
    _Float16* wn1t = (_Float16*)(ws + woff);  woff += (size_t)N_LAYERS * HID * 2 * HID * 2;
    _Float16* wn2t = (_Float16*)(ws + woff);

    // weight conversion (cheap; deterministic every call)
    prep_edge_w1<<<(N_LAYERS * HID * K1_PAD + 255) / 256, 256, 0, stream>>>(ew1, we1t);
    prep_node_w1<<<(N_LAYERS * HID * 2 * HID + 255) / 256, 256, 0, stream>>>(nw1, wn1t);
    prep_sq<<<(N_LAYERS * HID * HID + 255) / 256, 256, 0, stream>>>(ew2, we2t);
    prep_sq<<<(N_LAYERS * HID * HID + 255) / 256, 256, 0, stream>>>(nw2, wn2t);

    // embedding
    embed_kernel<<<(int)((NH + 255) / 256), 256, 0, stream>>>(nodes, emb_w, emb_b, hA);

    const float* hin = hA;
    float* hout = hB;
    for (int l = 0; l < N_LAYERS; ++l) {
        hipMemsetAsync(agg, 0, NH * 4, stream);
        edge_kernel<<<N_EDGES / 64, 128, 0, stream>>>(
            hin, eidx, emask, eattr,
            we1t + (size_t)l * HID * K1_PAD, eb1 + l * HID,
            we2t + (size_t)l * HID * HID,    eb2 + l * HID, agg);
        float* dst = (l == N_LAYERS - 1) ? (float*)d_out : hout;
        node_kernel<<<(N_NODES + 63) / 64, 128, 0, stream>>>(
            hin, agg,
            wn1t + (size_t)l * HID * 2 * HID, nb1 + l * HID,
            wn2t + (size_t)l * HID * HID,     nb2 + l * HID, dst);
        const float* t = hin;
        hin = dst;
        hout = (float*)t;
    }
}
